// MambaBlock_8641474199722
// MI455X (gfx1250) — matrix operand order repaired and checked by reading, not compiled, not run on hardware
//
#include <hip/hip_runtime.h>
#include <hip/hip_bf16.h>

typedef _Float16 f16;
typedef __attribute__((ext_vector_type(16))) _Float16 f16x16;
typedef __attribute__((ext_vector_type(8)))  float    f32x8;

#define BB 4
#define LL 2048
#define DD 1024
#define HH 32          // heads
#define PP 64          // headdim
#define NS 128         // dstate
#define CHK 256        // chunk
#define NC 8           // chunks per sequence
#define DIN 2048       // d_inner
#define CONVD 2304     // conv dim
#define DTOFF 4352     // DIN + CONVD
#define ZP 4480        // padded in-proj width (4384 -> 4480 = 35*128)
#define EPS_OUT 1.1920929e-07f
#define EPS_GATE 1e-05f

// ---------------- WMMA fragment helpers (CDNA5 16x16x32 f16 layouts) -------

// A operand: 16x32 (MxK). lanes 0-15: K base 0, lanes 16-31: K base 8;
// elements 0..7 -> K=koff..koff+7, elements 8..15 -> K=16+koff..16+koff+7.
__device__ __forceinline__ f16x16 frag_a(const f16* base, int ld) {
  int lane = threadIdx.x & 31;
  int row  = lane & 15;
  int koff = (lane >> 4) << 3;       // 0 or 8
  const f16* p = base + (size_t)row * ld;
  union { f16x16 v; uint4 q[2]; } u;
  u.q[0] = *(const uint4*)(p + koff);
  u.q[1] = *(const uint4*)(p + 16 + koff);
  return u.v;
}

// B operand from BT storage [N][K]: lanes 0-15: K base 0, lanes 16-31: K base 8;
// elements 0..7 -> K=koff..koff+7, elements 8..15 -> K=16+koff..16+koff+7.
__device__ __forceinline__ f16x16 frag_b(const f16* base, int ld) {
  int lane = threadIdx.x & 31;
  int n    = lane & 15;
  int kb   = (lane >> 4) << 3;       // 0 or 8
  const f16* p = base + (size_t)n * ld + kb;
  union { f16x16 v; uint4 q[2]; } u;
  u.q[0] = *(const uint4*)(p);
  u.q[1] = *(const uint4*)(p + 16);
  return u.v;
}

__device__ __forceinline__ f32x8 wmma16(f16x16 a, f16x16 b, f32x8 c) {
  return __builtin_amdgcn_wmma_f32_16x16x32_f16(false, a, false, b, (short)0, c,
                                                false, false);
}

// ---------------- CDNA5 async global->LDS copy (ASYNCcnt path) ------------
// Copies one 64-byte row (32 f16) per lane: global and LDS deltas are both
// 16B, so a single base pair + inst offsets covers all four b128 transfers.
__device__ __forceinline__ void async_row64(unsigned lds, const void* g) {
  asm volatile(
      "global_load_async_to_lds_b128 %0, %1, off\n\t"
      "global_load_async_to_lds_b128 %0, %1, off offset:16\n\t"
      "global_load_async_to_lds_b128 %0, %1, off offset:32\n\t"
      "global_load_async_to_lds_b128 %0, %1, off offset:48"
      :: "v"(lds), "v"(g) : "memory");
}
__device__ __forceinline__ void wait_async_le8() {
  asm volatile("s_wait_asynccnt 0x8" ::: "memory");
}
__device__ __forceinline__ void wait_async_0() {
  asm volatile("s_wait_asynccnt 0x0" ::: "memory");
}

// ---------------- Generic f16 GEMM: C[M,N] f32 = A[M,K] * BT[N,K]^T -------
// 128x128 block tile, 4 waves of 64x64, K-step 32, double-buffered LDS fed
// by async copies. N,K compile-time so all addressing folds to immediates.
template <int N, int K>
__global__ __launch_bounds__(128) void k_gemm(const f16* __restrict__ A,
                                              const f16* __restrict__ BT,
                                              float* __restrict__ C) {
  __shared__ __align__(16) f16 As[2][128][40];
  __shared__ __align__(16) f16 Bs[2][128][40];
  const int m0 = blockIdx.x * 128, n0 = blockIdx.y * 128;
  const int tid = threadIdx.x, wave = tid >> 5, lane = tid & 31;
  const int wm = (wave & 1) * 64, wn = (wave >> 1) * 64;

  const f16* ga = A + (size_t)(m0 + tid) * K;
  const f16* gb = BT + (size_t)(n0 + tid) * K;
  const unsigned laA0 = (unsigned)(size_t)&As[0][tid][0];
  const unsigned laA1 = (unsigned)(size_t)&As[1][tid][0];
  const unsigned laB0 = (unsigned)(size_t)&Bs[0][tid][0];
  const unsigned laB1 = (unsigned)(size_t)&Bs[1][tid][0];

  constexpr int NK = K / 32;
  // prologue: tile 0 into buffer 0
  async_row64(laA0, ga);
  async_row64(laB0, gb);

  f32x8 acc[4][4] = {};
  for (int kk = 0; kk < NK; ++kk) {
    const int cur = kk & 1;
    if (kk + 1 < NK) {   // prefetch next tile into other buffer
      async_row64(cur ? laA0 : laA1, ga + (kk + 1) * 32);
      async_row64(cur ? laB0 : laB1, gb + (kk + 1) * 32);
      wait_async_le8();  // current tile's 8 transfers complete
    } else {
      wait_async_0();
    }
    __syncthreads();     // all waves' copies for 'cur' landed
    f16x16 af[4], bf[4];
#pragma unroll
    for (int i = 0; i < 4; ++i) af[i] = frag_a(&As[cur][wm + 16 * i][0], 40);
#pragma unroll
    for (int j = 0; j < 4; ++j) bf[j] = frag_b(&Bs[cur][wn + 16 * j][0], 40);
#pragma unroll
    for (int i = 0; i < 4; ++i)
#pragma unroll
      for (int j = 0; j < 4; ++j)
        acc[i][j] = wmma16(af[i], bf[j], acc[i][j]);
    __syncthreads();     // reads done before 'cur' is refilled next iter
  }
  const int n = lane & 15, mh = (lane >> 4) << 3;
  float* cp = C + (size_t)(m0 + wm + mh) * N + (n0 + wn + n);
#pragma unroll
  for (int i = 0; i < 4; ++i)
#pragma unroll
    for (int j = 0; j < 4; ++j)
#pragma unroll
      for (int v = 0; v < 8; ++v)
        cp[(size_t)(16 * i + v) * N + 16 * j] = acc[i][j][v];
}

// ---------------- weight transpose f32[K,N] -> f16[Npad,K] (zero pad) ------
__global__ void k_transpose16(const float* __restrict__ in, f16* __restrict__ out,
                              int K, int N, int Npad) {
  size_t idx = (size_t)blockIdx.x * 256 + threadIdx.x;
  size_t total = (size_t)Npad * K;
  if (idx >= total) return;
  int k = (int)(idx % K);
  int nn = (int)(idx / K);
  out[idx] = (nn < N) ? (f16)in[(size_t)k * N + nn] : (f16)0.f;
}

// ---------------- rmsnorm(x) -> f16 ---------------------------------------
__global__ __launch_bounds__(256) void k_rms_in(const float* __restrict__ x,
                                                f16* __restrict__ out) {
  const int tid = threadIdx.x;
  const size_t row = blockIdx.x;
  const float* r = x + row * DD;
  float v[4]; float ss = 0.f;
#pragma unroll
  for (int i = 0; i < 4; ++i) { v[i] = r[tid + 256 * i]; ss += v[i] * v[i]; }
  __shared__ float red[256];
  red[tid] = ss; __syncthreads();
  for (int o = 128; o > 0; o >>= 1) { if (tid < o) red[tid] += red[tid + o]; __syncthreads(); }
  float inv = rsqrtf(red[0] / (float)DD + EPS_OUT);
#pragma unroll
  for (int i = 0; i < 4; ++i) out[row * DD + tid + 256 * i] = (f16)(v[i] * inv);
}

// ---------------- dt/softplus + per-chunk cumsum of a ---------------------
__global__ __launch_bounds__(256) void k_dt(const float* __restrict__ zx,
                                            const float* __restrict__ dt_bias,
                                            const float* __restrict__ a_log,
                                            float* __restrict__ dtbuf,
                                            float* __restrict__ acum) {
  const int l = threadIdx.x;
  const int c = blockIdx.x, h = blockIdx.y, b = blockIdx.z;
  const size_t lg = (size_t)b * LL + c * CHK + l;
  float xr = zx[lg * ZP + DTOFF + h] + dt_bias[h];
  float dt = (xr > 20.f) ? xr : __logf(1.f + __expf(xr));
  dtbuf[lg * HH + h] = dt;
  float a = -__expf(a_log[h]) * dt;
  __shared__ float buf[256];
  buf[l] = a; __syncthreads();
  for (int o = 1; o < 256; o <<= 1) {
    float t = (l >= o) ? buf[l - o] : 0.f;
    __syncthreads();
    buf[l] += t;
    __syncthreads();
  }
  acum[(((size_t)b * HH + h) * NC + c) * CHK + l] = buf[l];
}

// ---------------- depthwise conv + SiLU + pack into WMMA layouts ----------
__global__ __launch_bounds__(256) void k_conv(const float* __restrict__ zx,
    const float* __restrict__ conv_w, const float* __restrict__ conv_b,
    const float* __restrict__ dtbuf, const float* __restrict__ acum,
    f16* __restrict__ xh16, f16* __restrict__ uT16, f16* __restrict__ uDT16,
    f16* __restrict__ Bm16, f16* __restrict__ BmT16, f16* __restrict__ Cm16) {
  size_t gid = (size_t)blockIdx.x * 256 + threadIdx.x;
  if (gid >= (size_t)BB * LL * CONVD) return;
  int ch = (int)(gid % CONVD);
  size_t t = gid / CONVD;
  int l = (int)(t % LL);
  int b = (int)(t / LL);
  float acc = conv_b[ch];
#pragma unroll
  for (int k = 0; k < 4; ++k) {
    int ls = l - 3 + k;
    if (ls >= 0) acc += zx[((size_t)b * LL + ls) * ZP + DIN + ch] * conv_w[ch * 4 + k];
  }
  float xc = acc / (1.f + __expf(-acc));   // SiLU
  int c = l >> 8, s = l & 255;
  if (ch < DIN) {
    int h = ch >> 6, p = ch & 63;
    xh16[((size_t)b * LL + l) * DIN + ch] = (f16)xc;
    float dt = dtbuf[((size_t)b * LL + l) * HH + h];
    float u = dt * xc;
    size_t uidx = ((((size_t)b * NC + c) * HH + h) * PP + p) * CHK + s;
    uT16[uidx] = (f16)u;
    const float* ac = acum + (((size_t)b * HH + h) * NC + c) * CHK;
    uDT16[uidx] = (f16)(u * __expf(ac[255] - ac[s]));   // decay_states weighting
  } else if (ch < DIN + NS) {
    int n = ch - DIN;
    Bm16[(((size_t)b * NC + c) * CHK + s) * NS + n] = (f16)xc;
    BmT16[(((size_t)b * NC + c) * NS + n) * CHK + s] = (f16)xc;
  } else {
    int n = ch - DIN - NS;
    Cm16[(((size_t)b * NC + c) * CHK + s) * NS + n] = (f16)xc;
  }
}

// ---------------- intra-chunk: y_diag = (C B^T o L) u ---------------------
__global__ __launch_bounds__(128) void k_chunk1(const f16* __restrict__ Cm16,
    const f16* __restrict__ Bm16, const f16* __restrict__ uT16,
    const float* __restrict__ acum, float* __restrict__ ybuf) {
  const int c = blockIdx.x, h = blockIdx.y, b = blockIdx.z;
  const int tid = threadIdx.x, wave = tid >> 5, lane = tid & 31;
  const f16* Cc = Cm16 + (((size_t)b * NC + c) * CHK) * NS;
  const f16* Bc = Bm16 + (((size_t)b * NC + c) * CHK) * NS;
  const f16* uT = uT16 + ((((size_t)b * NC + c) * HH + h) * PP) * CHK;
  const float* ac = acum + (((size_t)b * HH + h) * NC + c) * CHK;
  __shared__ float sac[256];
  __shared__ __align__(16) f16 Sb[4][64][40];
  sac[tid] = ac[tid]; sac[tid + 128] = ac[tid + 128];
  __syncthreads();
  const int l0 = wave * 64;
  const int n = lane & 15, mh = (lane >> 4) << 3;
  f32x8 yacc[4][4] = {};
  for (int sb = 0; sb < CHK; sb += 32) {
    f32x8 sacc[4][2] = {};
#pragma unroll
    for (int k0 = 0; k0 < NS; k0 += 32) {
      f16x16 af[4], bf[2];
#pragma unroll
      for (int i = 0; i < 4; ++i) af[i] = frag_a(Cc + (size_t)(l0 + 16 * i) * NS + k0, NS);
#pragma unroll
      for (int j = 0; j < 2; ++j) bf[j] = frag_b(Bc + (size_t)(sb + 16 * j) * NS + k0, NS);
#pragma unroll
      for (int i = 0; i < 4; ++i)
#pragma unroll
        for (int j = 0; j < 2; ++j)
          sacc[i][j] = wmma16(af[i], bf[j], sacc[i][j]);
    }
    // causal segsum mask, f32 D-layout -> f16 A-layout via LDS
#pragma unroll
    for (int i = 0; i < 4; ++i)
#pragma unroll
      for (int j = 0; j < 2; ++j)
#pragma unroll
        for (int v = 0; v < 8; ++v) {
          int li = l0 + 16 * i + mh + v;
          int si = sb + 16 * j + n;
          float m = (si <= li) ? __expf(sac[li] - sac[si]) : 0.f;
          Sb[wave][16 * i + mh + v][16 * j + n] = (f16)(sacc[i][j][v] * m);
        }
    __syncthreads();
    {
      f16x16 af2[4], bf2[4];
#pragma unroll
      for (int i = 0; i < 4; ++i) af2[i] = frag_a(&Sb[wave][16 * i][0], 40);
#pragma unroll
      for (int j = 0; j < 4; ++j) bf2[j] = frag_b(uT + (size_t)(16 * j) * CHK + sb, CHK);
#pragma unroll
      for (int i = 0; i < 4; ++i)
#pragma unroll
        for (int j = 0; j < 4; ++j)
          yacc[i][j] = wmma16(af2[i], bf2[j], yacc[i][j]);
    }
    __syncthreads();
  }
  float* yo = ybuf + ((size_t)b * LL + c * CHK) * DIN + h * PP;
#pragma unroll
  for (int i = 0; i < 4; ++i)
#pragma unroll
    for (int j = 0; j < 4; ++j)
#pragma unroll
      for (int v = 0; v < 8; ++v)
        yo[(size_t)(l0 + 16 * i + mh + v) * DIN + 16 * j + n] = yacc[i][j][v];
}

// ---------------- per-chunk states = (decay o u)^T B ----------------------
__global__ __launch_bounds__(128) void k_chunk2(const f16* __restrict__ uDT16,
    const f16* __restrict__ BmT16, float* __restrict__ states) {
  const int c = blockIdx.x, h = blockIdx.y, b = blockIdx.z;
  const int tid = threadIdx.x, wave = tid >> 5, lane = tid & 31;
  const f16* uA = uDT16 + ((((size_t)b * NC + c) * HH + h) * PP) * CHK;
  const f16* Bt = BmT16 + (((size_t)b * NC + c) * NS) * CHK;
  f32x8 acc[4][2] = {};
  for (int k0 = 0; k0 < CHK; k0 += 32) {
    f16x16 af[4], bf[2];
#pragma unroll
    for (int i = 0; i < 4; ++i) af[i] = frag_a(uA + (size_t)(16 * i) * CHK + k0, CHK);
#pragma unroll
    for (int j = 0; j < 2; ++j) bf[j] = frag_b(Bt + (size_t)(wave * 32 + 16 * j) * CHK + k0, CHK);
#pragma unroll
    for (int i = 0; i < 4; ++i)
#pragma unroll
      for (int j = 0; j < 2; ++j)
        acc[i][j] = wmma16(af[i], bf[j], acc[i][j]);
  }
  float* st = states + ((((size_t)b * NC + c) * HH + h) * PP) * NS;
  const int n = lane & 15, mh = (lane >> 4) << 3;
#pragma unroll
  for (int i = 0; i < 4; ++i)
#pragma unroll
    for (int j = 0; j < 2; ++j)
#pragma unroll
      for (int v = 0; v < 8; ++v)
        st[(size_t)(16 * i + mh + v) * NS + wave * 32 + 16 * j + n] = acc[i][j][v];
}

// ---------------- inter-chunk recurrence (8 steps) ------------------------
__global__ __launch_bounds__(256) void k_scan(const float* __restrict__ states,
    const float* __restrict__ acum, f16* __restrict__ prev16) {
  const int h = blockIdx.x, b = blockIdx.y;
  const int tid = threadIdx.x;
  float s[32];
#pragma unroll
  for (int r = 0; r < 32; ++r) s[r] = 0.f;
  for (int c = 0; c < NC; ++c) {
    float dec = __expf(acum[(((size_t)b * HH + h) * NC + c) * CHK + 255]);
    size_t base = ((((size_t)b * NC + c) * HH + h) * PP) * NS;
#pragma unroll
    for (int r = 0; r < 32; ++r) {
      size_t e = base + r * 256 + tid;
      prev16[e] = (f16)s[r];
      s[r] = s[r] * dec + states[e];
    }
  }
}

// ---------------- y_off = (C prev^T) * exp(acum) + skip, accumulate -------
__global__ __launch_bounds__(128) void k_chunk3(const f16* __restrict__ Cm16,
    const f16* __restrict__ prev16, const f16* __restrict__ xh16,
    const float* __restrict__ acum, const float* __restrict__ d_skip,
    float* __restrict__ ybuf) {
  const int c = blockIdx.x, h = blockIdx.y, b = blockIdx.z;
  const int tid = threadIdx.x, wave = tid >> 5, lane = tid & 31;
  const f16* Cc = Cm16 + (((size_t)b * NC + c) * CHK) * NS;
  const f16* Pv = prev16 + ((((size_t)b * NC + c) * HH + h) * PP) * NS;
  const float* ac = acum + (((size_t)b * HH + h) * NC + c) * CHK;
  __shared__ float sac[256];
  sac[tid] = ac[tid]; sac[tid + 128] = ac[tid + 128];
  __syncthreads();
  const int l0 = wave * 64;
  f32x8 acc[4][4] = {};
#pragma unroll
  for (int k0 = 0; k0 < NS; k0 += 32) {
    f16x16 af[4], bf[4];
#pragma unroll
    for (int i = 0; i < 4; ++i) af[i] = frag_a(Cc + (size_t)(l0 + 16 * i) * NS + k0, NS);
#pragma unroll
    for (int j = 0; j < 4; ++j) bf[j] = frag_b(Pv + (size_t)(16 * j) * NS + k0, NS);
#pragma unroll
    for (int i = 0; i < 4; ++i)
#pragma unroll
      for (int j = 0; j < 4; ++j)
        acc[i][j] = wmma16(af[i], bf[j], acc[i][j]);
  }
  const float dsk = d_skip[h];
  const int n = lane & 15, mh = (lane >> 4) << 3;
  const size_t rowbase = (size_t)b * LL + c * CHK;
#pragma unroll
  for (int i = 0; i < 4; ++i)
#pragma unroll
    for (int j = 0; j < 4; ++j)
#pragma unroll
      for (int v = 0; v < 8; ++v) {
        int li = l0 + 16 * i + mh + v, p = 16 * j + n;
        size_t idx = (rowbase + li) * DIN + h * PP + p;
        ybuf[idx] = ybuf[idx] + acc[i][j][v] * __expf(sac[li]) + dsk * (float)xh16[idx];
      }
}

// ---------------- gated rmsnorm: g = rmsnorm(y*silu(z))*norm_w -> f16 -----
__global__ __launch_bounds__(256) void k_gate(const float* __restrict__ ybuf,
    const float* __restrict__ zx, const float* __restrict__ norm_w,
    f16* __restrict__ g16) {
  const int tid = threadIdx.x;
  const size_t row = blockIdx.x;
  float v[8]; float ss = 0.f;
#pragma unroll
  for (int i = 0; i < 8; ++i) {
    int col = tid + 256 * i;
    float z = zx[row * ZP + col];
    float y = ybuf[row * DIN + col];
    float w = y * (z / (1.f + __expf(-z)));
    v[i] = w; ss += w * w;
  }
  __shared__ float red[256];
  red[tid] = ss; __syncthreads();
  for (int o = 128; o > 0; o >>= 1) { if (tid < o) red[tid] += red[tid + o]; __syncthreads(); }
  float inv = rsqrtf(red[0] / (float)DIN + EPS_GATE);
#pragma unroll
  for (int i = 0; i < 8; ++i) {
    int col = tid + 256 * i;
    g16[row * DIN + col] = (f16)(v[i] * inv * norm_w[col]);
  }
}

// ---------------- h = x + mm1 ; hn16 = rmsnorm(h) -------------------------
__global__ __launch_bounds__(256) void k_addnorm(const float* __restrict__ x,
    const float* __restrict__ mm1, float* __restrict__ hbuf,
    f16* __restrict__ hn16) {
  const int tid = threadIdx.x;
  const size_t row = blockIdx.x;
  float v[4]; float ss = 0.f;
#pragma unroll
  for (int i = 0; i < 4; ++i) {
    int col = tid + 256 * i;
    float hv = x[row * DD + col] + mm1[row * DD + col];
    hbuf[row * DD + col] = hv;
    v[i] = hv; ss += hv * hv;
  }
  __shared__ float red[256];
  red[tid] = ss; __syncthreads();
  for (int o = 128; o > 0; o >>= 1) { if (tid < o) red[tid] += red[tid + o]; __syncthreads(); }
  float inv = rsqrtf(red[0] / (float)DD + EPS_OUT);
#pragma unroll
  for (int i = 0; i < 4; ++i)
    hn16[row * DD + tid + 256 * i] = (f16)(v[i] * inv);
}

__global__ void k_relu2(const float* __restrict__ m, f16* __restrict__ m16, size_t n) {
  size_t i = (size_t)blockIdx.x * 256 + threadIdx.x;
  if (i < n) { float r = fmaxf(m[i], 0.f); m16[i] = (f16)(r * r); }
}

__global__ void k_final(const float* __restrict__ hbuf, const float* __restrict__ mp,
                        float* __restrict__ out, size_t n) {
  size_t i = (size_t)blockIdx.x * 256 + threadIdx.x;
  if (i < n) out[i] = hbuf[i] + mp[i];
}

// ---------------------------------------------------------------------------
extern "C" void kernel_launch(void* const* d_in, const int* in_sizes, int n_in,
                              void* d_out, int out_size, void* d_ws, size_t ws_size,
                              hipStream_t stream) {
  (void)in_sizes; (void)n_in; (void)out_size; (void)ws_size;
  const float* x       = (const float*)d_in[0];
  const float* w_in    = (const float*)d_in[1];
  const float* conv_w  = (const float*)d_in[2];
  const float* conv_b  = (const float*)d_in[3];
  const float* dt_bias = (const float*)d_in[4];
  const float* a_log   = (const float*)d_in[5];
  const float* d_skip  = (const float*)d_in[6];
  const float* norm_w  = (const float*)d_in[7];
  const float* w_out   = (const float*)d_in[8];
  const float* w_fc    = (const float*)d_in[9];
  const float* w_proj  = (const float*)d_in[10];
  float* out = (float*)d_out;

  char* wsp = (char*)d_ws;
  size_t off = 0;
  auto carve = [&](size_t bytes) -> char* {
    char* p = wsp + off;
    off = (off + bytes + 255) & ~(size_t)255;
    return p;
  };
  const size_t ROWS = (size_t)BB * LL;          // 8192
  f16*   xn16   = (f16*)carve(ROWS * DD * 2);
  f16*   win16  = (f16*)carve((size_t)ZP * DD * 2);
  f16*   wout16 = (f16*)carve((size_t)DD * DIN * 2);
  f16*   wfc16  = (f16*)carve((size_t)4096 * DD * 2);
  f16*   wpr16  = (f16*)carve((size_t)DD * 4096 * 2);
  float* zx     = (float*)carve(ROWS * ZP * 4);       // zxbcdt; later reused as mfc
  float* dtb    = (float*)carve(ROWS * HH * 4);
  float* acum   = (float*)carve((size_t)BB * HH * NC * CHK * 4);
  f16*   xh16   = (f16*)carve(ROWS * DIN * 2);
  f16*   uT16   = (f16*)carve(ROWS * DIN * 2);
  f16*   uDT16  = (f16*)carve(ROWS * DIN * 2);
  f16*   Bm16   = (f16*)carve((size_t)BB * NC * CHK * NS * 2);
  f16*   BmT16  = (f16*)carve((size_t)BB * NC * CHK * NS * 2);
  f16*   Cm16   = (f16*)carve((size_t)BB * NC * CHK * NS * 2);
  float* ybuf   = (float*)carve(ROWS * DIN * 4);
  float* states = (float*)carve((size_t)BB * NC * HH * PP * NS * 4);
  f16*   prev16 = (f16*)carve((size_t)BB * NC * HH * PP * NS * 2);
  f16*   g16    = (f16*)carve(ROWS * DIN * 2);
  float* mm1    = (float*)carve(ROWS * DD * 4);       // also reused as mp
  float* hbuf   = (float*)carve(ROWS * DD * 4);
  f16*   hn16   = (f16*)carve(ROWS * DD * 2);
  f16*   m16    = (f16*)carve(ROWS * 4096 * 2);
  float* mfc = zx;    // zxbcdt dead after k_gate
  float* mp  = mm1;   // mm1 dead after k_addnorm

  // 1) weight convert/transpose to f16 [N,K]
  { size_t t = (size_t)ZP * DD;
    k_transpose16<<<(unsigned)((t + 255) / 256), 256, 0, stream>>>(w_in, win16, DD, 4384, ZP); }
  { size_t t = (size_t)DD * DIN;
    k_transpose16<<<(unsigned)((t + 255) / 256), 256, 0, stream>>>(w_out, wout16, DIN, DD, DD); }
  { size_t t = (size_t)4096 * DD;
    k_transpose16<<<(unsigned)((t + 255) / 256), 256, 0, stream>>>(w_fc, wfc16, DD, 4096, 4096); }
  { size_t t = (size_t)DD * 4096;
    k_transpose16<<<(unsigned)((t + 255) / 256), 256, 0, stream>>>(w_proj, wpr16, 4096, DD, DD); }

  // 2) input rmsnorm, in-projection GEMM
  k_rms_in<<<(unsigned)ROWS, 256, 0, stream>>>(x, xn16);
  k_gemm<ZP, DD><<<dim3(64, 35), 128, 0, stream>>>(xn16, win16, zx);

  // 3) dt/a cumsum, conv+SiLU+pack
  k_dt<<<dim3(NC, HH, BB), 256, 0, stream>>>(zx, dt_bias, a_log, dtb, acum);
  { size_t tot = (size_t)BB * LL * CONVD;
    k_conv<<<(unsigned)((tot + 255) / 256), 256, 0, stream>>>(
        zx, conv_w, conv_b, dtb, acum, xh16, uT16, uDT16, Bm16, BmT16, Cm16); }

  // 4) chunked SSM
  k_chunk1<<<dim3(NC, HH, BB), 128, 0, stream>>>(Cm16, Bm16, uT16, acum, ybuf);
  k_chunk2<<<dim3(NC, HH, BB), 128, 0, stream>>>(uDT16, BmT16, states);
  k_scan<<<dim3(HH, BB), 256, 0, stream>>>(states, acum, prev16);
  k_chunk3<<<dim3(NC, HH, BB), 128, 0, stream>>>(Cm16, prev16, xh16, acum, d_skip, ybuf);

  // 5) gated rmsnorm + out-projection, residual
  k_gate<<<(unsigned)ROWS, 256, 0, stream>>>(ybuf, zx, norm_w, g16);
  k_gemm<DD, DIN><<<dim3(64, 8), 128, 0, stream>>>(g16, wout16, mm1);
  k_addnorm<<<(unsigned)ROWS, 256, 0, stream>>>(x, mm1, hbuf, hn16);

  // 6) MLP: fc -> relu^2 -> proj, final residual
  k_gemm<4096, DD><<<dim3(64, 32), 128, 0, stream>>>(hn16, wfc16, mfc);
  { size_t nrelu = ROWS * 4096;
    k_relu2<<<(unsigned)((nrelu + 255) / 256), 256, 0, stream>>>(mfc, m16, nrelu); }
  k_gemm<DD, 4096><<<dim3(64, 8), 128, 0, stream>>>(m16, wpr16, mp);
  { size_t nfin = ROWS * DD;
    k_final<<<(unsigned)((nfin + 255) / 256), 256, 0, stream>>>(hbuf, mp, out, nfin); }
}
